// CS_ELMO_14181982012016
// MI455X (gfx1250) — matrix operand order repaired and checked by reading, not compiled, not run on hardware
//
#include <hip/hip_runtime.h>
#include <hip/hip_bf16.h>
#include <math.h>

// ---------------- problem constants ----------------
#define Bn   64
#define Wn   128
#define Cn   20
#define Dn   256
#define Hn   512
#define Gn   2048
#define NPOS (Bn*Wn*Cn)      // 163840 conv positions
#define NW   (Bn*Wn)         // 8192 words

typedef __attribute__((ext_vector_type(16))) __bf16 v16bf;
typedef __attribute__((ext_vector_type(8)))  float  v8f;

// ---------------- bf16 helpers ----------------
__device__ __forceinline__ __bf16 f2bf(float f) {
  union { float f; unsigned u; } v; v.f = f;
  unsigned u = v.u + 0x7FFFu + ((v.u >> 16) & 1u);
  union { unsigned short s; __bf16 b; } o; o.s = (unsigned short)(u >> 16);
  return o.b;
}
__device__ __forceinline__ float bf2f(__bf16 b) {
  union { unsigned short s; __bf16 b; } i; i.b = b;
  union { unsigned u; float f; } o; o.u = ((unsigned)i.s) << 16;
  return o.f;
}
__device__ __forceinline__ float sigm(float x) { return 1.f / (1.f + __expf(-x)); }

// ---------------- WMMA fragment helpers (wave32; contiguous k order, same for A and B) ----------------
// A 16x32 bf16, row-major source: lane holds row M=lane%16; half=lane/16.
// lanes 0-15: K = k0+[0..15]; lanes 16-31: K = k0+16+[0..15].
__device__ __forceinline__ v16bf load_a_row(const __bf16* rowp, int half, int k0) {
  v16bf a;
  int kb = k0 + 16*half;
#pragma unroll
  for (int e = 0; e < 16; ++e) a[e]     = rowp[kb + e];
  return a;
}
// B 32x16 bf16 from an N-major (transposed) weight: lane holds col N=lane%16,
// elements are K = k0 + 16*half + [0..15] -> fully contiguous 32B per lane.
__device__ __forceinline__ v16bf load_bt_row(const __bf16* rowp, int half, int k0) {
  v16bf b;
  int kb = k0 + 16*half;
#pragma unroll
  for (int e = 0; e < 16; ++e) b[e] = rowp[kb + e];
  return b;
}
__device__ __forceinline__ v8f wmma_bf(v16bf a, v16bf b, v8f c) {
  return __builtin_amdgcn_wmma_f32_16x16x32_bf16(false, a, false, b, (short)0, c, false, false);
}
__device__ __forceinline__ v8f vzero8() { v8f z = {0.f,0.f,0.f,0.f,0.f,0.f,0.f,0.f}; return z; }

// ---------------- weight conversion kernels ----------------
// conv weight (D_out, D_in, taps) f32 -> N-major BT[o][t*Dn+i] bf16 (ld = taps*Dn)
__global__ void k_conv_w(const float* __restrict__ src, __bf16* __restrict__ dst, int taps) {
  int idx = blockIdx.x * blockDim.x + threadIdx.x;
  int total = Dn * Dn * taps;
  if (idx >= total) return;
  int o = idx / (Dn * taps);
  int rem = idx % (Dn * taps);
  int i = rem / taps;
  int t = rem % taps;
  dst[(size_t)o * (taps * Dn) + t * Dn + i] = f2bf(src[idx]);
}
// (R,C) f32 -> bf16; trans=1 produces (C,R) N-major layout
__global__ void k_mat_bf(const float* __restrict__ src, __bf16* __restrict__ dst,
                         int R, int C, int trans) {
  int idx = blockIdx.x * blockDim.x + threadIdx.x;
  if (idx >= R * C) return;
  int r = idx / C, c = idx % C;
  if (trans) dst[(size_t)c * R + r] = f2bf(src[idx]);
  else       dst[idx] = f2bf(src[idx]);
}

// ---------------- char embedding gather (zero-padded to C+2) ----------------
__global__ void k_embed(const int* __restrict__ src, const float* __restrict__ tab,
                        __bf16* __restrict__ xpad) {
  int row = blockIdx.x;            // 0 .. NW*(Cn+2)-1
  int d = threadIdx.x;             // 256
  int p = row / (Cn + 2), c = row % (Cn + 2);
  float v = 0.f;
  if (c < Cn) v = tab[(size_t)src[p * Cn + c] * Dn + d];
  xpad[(size_t)row * Dn + d] = f2bf(v);
}

// ---------------- conv-as-GEMM, 2x2 register-blocked (+bias +PE) ----------------
__global__ void k_conv_gemm(const __bf16* __restrict__ xpad, const __bf16* __restrict__ BT,
                            const float* __restrict__ bias, int taps,
                            __bf16* __restrict__ cat, int coloff) {
  int wave = threadIdx.x >> 5, lane = threadIdx.x & 31;
  int half = lane >> 4, lcol = lane & 15;
  int m0 = blockIdx.x * 32;                    // 32-row block (NPOS/32 blocks)
  int n0 = wave * 32;                          // 8 waves cover all 256 cols
  int Kld = taps * Dn;
  int P0 = m0 + (lane & 15), P1 = P0 + 16;     // A rows for the two M sub-tiles
  int nw0 = P0 / Cn, c0 = P0 % Cn;
  int nw1 = P1 / Cn, c1 = P1 % Cn;
  const __bf16* bt0 = BT + (size_t)(n0 + lcol) * Kld;
  const __bf16* bt1 = bt0 + (size_t)16 * Kld;
  v8f acc00 = vzero8(), acc01 = vzero8(), acc10 = vzero8(), acc11 = vzero8();
  for (int t = 0; t < taps; ++t) {
    const __bf16* ar0 = xpad + (size_t)(nw0 * (Cn + 2) + c0 + t) * Dn;
    const __bf16* ar1 = xpad + (size_t)(nw1 * (Cn + 2) + c1 + t) * Dn;
    const __bf16* b0p = bt0 + t * Dn;
    const __bf16* b1p = bt1 + t * Dn;
    for (int k0 = 0; k0 < Dn; k0 += 32) {
      __builtin_prefetch(ar0 + k0 + 64, 0, 1);
      v16bf a0 = load_a_row(ar0, half, k0);
      v16bf a1 = load_a_row(ar1, half, k0);
      v16bf b0 = load_bt_row(b0p, half, k0);
      v16bf b1 = load_bt_row(b1p, half, k0);
      acc00 = wmma_bf(a0, b0, acc00);
      acc01 = wmma_bf(a0, b1, acc01);
      acc10 = wmma_bf(a1, b0, acc10);
      acc11 = wmma_bf(a1, b1, acc11);
    }
  }
#pragma unroll
  for (int ct = 0; ct < 2; ++ct) {
    int nc = n0 + ct * 16 + lcol;
    float bia = bias[nc];
    float div = __expf((float)((nc >> 1) * 2) * (-9.210340371976184f / (float)Dn));
    int odd = nc & 1;
#pragma unroll
    for (int mt = 0; mt < 2; ++mt) {
      v8f acc = (mt == 0) ? (ct == 0 ? acc00 : acc01) : (ct == 0 ? acc10 : acc11);
#pragma unroll
      for (int r = 0; r < 8; ++r) {
        int Pr = m0 + mt * 16 + r + 8 * half;
        float ang = (float)(Pr % Cn) * div;
        float pe = odd ? __cosf(ang) : __sinf(ang);
        cat[(size_t)Pr * (2 * Dn) + coloff + nc] = f2bf(acc[r] + bia + pe);
      }
    }
  }
}

// ---------------- attention scores: s = tanh(cat@Wa + ba) . ua ----------------
__global__ void k_attn_score(const __bf16* __restrict__ cat, const __bf16* __restrict__ WaT,
                             const float* __restrict__ ba, const float* __restrict__ ua,
                             float* __restrict__ sout) {
  int wave = threadIdx.x >> 5, lane = threadIdx.x & 31;
  int half = lane >> 4, lcol = lane & 15;
  int mtile = blockIdx.x * 8 + wave;
  const __bf16* arow = cat + (size_t)(mtile * 16 + (lane & 15)) * (2 * Dn);
  float part[8];
#pragma unroll
  for (int r = 0; r < 8; ++r) part[r] = 0.f;
  for (int nt = 0; nt < (2 * Dn) / 16; ++nt) {
    int ncol = nt * 16 + lcol;
    const __bf16* btp = WaT + (size_t)ncol * (2 * Dn);
    v8f acc = vzero8();
    for (int k0 = 0; k0 < 2 * Dn; k0 += 32) {
      v16bf a = load_a_row(arow, half, k0);
      v16bf b = load_bt_row(btp, half, k0);
      acc = wmma_bf(a, b, acc);
    }
    float uan = ua[ncol], ban = ba[ncol];
#pragma unroll
    for (int r = 0; r < 8; ++r) part[r] += tanhf(acc[r] + ban) * uan;
  }
#pragma unroll
  for (int r = 0; r < 8; ++r) {
    float v = part[r];
    v += __shfl_xor(v, 1, 32);
    v += __shfl_xor(v, 2, 32);
    v += __shfl_xor(v, 4, 32);
    v += __shfl_xor(v, 8, 32);
    if (lcol == 0) sout[mtile * 16 + r + 8 * half] = v;
  }
}

// ---------------- softmax over chars + weighted sum ----------------
__global__ void k_attn_pool(const __bf16* __restrict__ cat, const float* __restrict__ s,
                            __bf16* __restrict__ wemb) {
  int n = blockIdx.x;
  int tid = threadIdx.x;                       // 128
  __shared__ float sm[Cn], aw[Cn], red[2];
  if (tid < Cn) sm[tid] = s[n * Cn + tid];
  __syncthreads();
  if (tid == 0) {
    float mx = sm[0];
    for (int c = 1; c < Cn; ++c) mx = fmaxf(mx, sm[c]);
    float den = 0.f;
    for (int c = 0; c < Cn; ++c) den += __expf(sm[c] - mx);
    red[0] = mx; red[1] = den;
  }
  __syncthreads();
  if (tid < Cn) aw[tid] = __expf(sm[tid] - red[0]) / red[1];
  __syncthreads();
  for (int d = tid; d < 2 * Dn; d += blockDim.x) {
    float acc = 0.f;
#pragma unroll
    for (int c = 0; c < Cn; ++c)
      acc += aw[c] * bf2f(cat[(size_t)(n * Cn + c) * (2 * Dn) + d]);
    wemb[(size_t)n * (2 * Dn) + d] = f2bf(acc);
  }
}

// ---------------- wemb @ W1 -> feats[:, 0:256], 2x2 blocked ----------------
__global__ void k_w1_gemm(const __bf16* __restrict__ wemb, const __bf16* __restrict__ W1T,
                          __bf16* __restrict__ feats) {
  int wave = threadIdx.x >> 5, lane = threadIdx.x & 31;
  int half = lane >> 4, lcol = lane & 15;
  int m0 = blockIdx.x * 32;                    // NW/32 blocks
  int n0 = wave * 32;                          // 8 waves cover 256 cols
  const __bf16* ar0 = wemb + (size_t)(m0 + (lane & 15)) * (2 * Dn);
  const __bf16* ar1 = ar0 + (size_t)16 * (2 * Dn);
  const __bf16* bt0 = W1T + (size_t)(n0 + lcol) * (2 * Dn);
  const __bf16* bt1 = bt0 + (size_t)16 * (2 * Dn);
  v8f acc00 = vzero8(), acc01 = vzero8(), acc10 = vzero8(), acc11 = vzero8();
  for (int k0 = 0; k0 < 2 * Dn; k0 += 32) {
    v16bf a0 = load_a_row(ar0, half, k0);
    v16bf a1 = load_a_row(ar1, half, k0);
    v16bf b0 = load_bt_row(bt0, half, k0);
    v16bf b1 = load_bt_row(bt1, half, k0);
    acc00 = wmma_bf(a0, b0, acc00);
    acc01 = wmma_bf(a0, b1, acc01);
    acc10 = wmma_bf(a1, b0, acc10);
    acc11 = wmma_bf(a1, b1, acc11);
  }
#pragma unroll
  for (int ct = 0; ct < 2; ++ct) {
    int nc = n0 + ct * 16 + lcol;
#pragma unroll
    for (int mt = 0; mt < 2; ++mt) {
      v8f acc = (mt == 0) ? (ct == 0 ? acc00 : acc01) : (ct == 0 ? acc10 : acc11);
#pragma unroll
      for (int r = 0; r < 8; ++r)
        feats[(size_t)(m0 + mt * 16 + r + 8 * half) * (2 * Dn) + nc] = f2bf(acc[r]);
    }
  }
}

// ---------------- word table gather -> feats[:, 256:512] ----------------
__global__ void k_word_gather(const int* __restrict__ wsrc, const float* __restrict__ wtab,
                              __bf16* __restrict__ feats) {
  int n = blockIdx.x, d = threadIdx.x;
  feats[(size_t)n * (2 * Dn) + Dn + d] = f2bf(wtab[(size_t)wsrc[n] * Dn + d]);
}

// ---------------- LSTM input projection, 2x2 blocked ----------------
// xg[t][b][g] = X[b][t] @ wih^T + bias ; wih already (N=G, K) row-major bf16.
__global__ void k_inproj(const __bf16* __restrict__ X, int Kdim, const __bf16* __restrict__ WT,
                         const float* __restrict__ bias, float* __restrict__ xg) {
  int wave = threadIdx.x >> 5, lane = threadIdx.x & 31;
  int half = lane >> 4, lcol = lane & 15;
  int m0 = blockIdx.x * 32;                          // NW/32 blocks
  int n0 = (blockIdx.y * 8 + wave) * 32;             // Gn/32 tiles
  int r0 = m0 + (lane & 15), r1 = r0 + 16;           // rows r = t*64+b
  const __bf16* ar0 = X + (size_t)((r0 & (Bn - 1)) * Wn + (r0 >> 6)) * Kdim;
  const __bf16* ar1 = X + (size_t)((r1 & (Bn - 1)) * Wn + (r1 >> 6)) * Kdim;
  const __bf16* bt0 = WT + (size_t)(n0 + lcol) * Kdim;
  const __bf16* bt1 = bt0 + (size_t)16 * Kdim;
  v8f acc00 = vzero8(), acc01 = vzero8(), acc10 = vzero8(), acc11 = vzero8();
  for (int k0 = 0; k0 < Kdim; k0 += 32) {
    __builtin_prefetch(ar0 + k0 + 64, 0, 1);
    v16bf a0 = load_a_row(ar0, half, k0);
    v16bf a1 = load_a_row(ar1, half, k0);
    v16bf b0 = load_bt_row(bt0, half, k0);
    v16bf b1 = load_bt_row(bt1, half, k0);
    acc00 = wmma_bf(a0, b0, acc00);
    acc01 = wmma_bf(a0, b1, acc01);
    acc10 = wmma_bf(a1, b0, acc10);
    acc11 = wmma_bf(a1, b1, acc11);
  }
#pragma unroll
  for (int ct = 0; ct < 2; ++ct) {
    int nc = n0 + ct * 16 + lcol;
    float bi = bias[nc];
#pragma unroll
    for (int mt = 0; mt < 2; ++mt) {
      v8f acc = (mt == 0) ? (ct == 0 ? acc00 : acc01) : (ct == 0 ? acc10 : acc11);
#pragma unroll
      for (int rr = 0; rr < 8; ++rr)
        xg[(size_t)(m0 + mt * 16 + rr + 8 * half) * Gn + nc] = acc[rr] + bi;
    }
  }
}

// ---------------- LSTM recurrence (one workgroup per direction) ----------------
// h resident in LDS (bf16, double-buffered), c in VGPRs; one barrier per step.
__global__ void __launch_bounds__(1024)
k_lstm(const float* __restrict__ xg_all, const __bf16* __restrict__ whh_all,
       __bf16* __restrict__ outfeat) {
  extern __shared__ __bf16 hb[];               // 2 * 64 * 512 bf16 = 128 KB
  int dir = blockIdx.x;
  const float*  xg  = xg_all  + (size_t)dir * Wn * Bn * Gn;
  const __bf16* whh = whh_all + (size_t)dir * Gn * Hn;   // (N=G, K=H) row-major
  int colOff = dir * Hn;
  int rev = dir;
  int tid = threadIdx.x;
  int wave = tid >> 5, lane = tid & 31, half = lane >> 4, lcol = lane & 15;

  for (int i = tid; i < 2 * Bn * Hn; i += blockDim.x) hb[i] = f2bf(0.f);
  float cst[4][8];
#pragma unroll
  for (int q = 0; q < 4; ++q)
#pragma unroll
    for (int r = 0; r < 8; ++r) cst[q][r] = 0.f;
  __syncthreads();

  for (int s = 0; s < Wn; ++s) {
    int t = rev ? (Wn - 1 - s) : s;
    const __bf16* hcur = hb + (size_t)(s & 1) * Bn * Hn;
    __bf16*       hnxt = hb + (size_t)((s & 1) ^ 1) * Bn * Hn;
    const float*  xgr  = xg + (size_t)t * Bn * Gn;
#pragma unroll
    for (int q = 0; q < 4; ++q) {
      int gidx = wave * 4 + q;                 // 128 tile-groups: mtile x jcol
      int mtile = gidx >> 5;                   // 0..3  (batch tiles)
      int j = gidx & 31;                       // 0..31 (h-column tiles)
      int nb = j * 16 + lcol;
      const __bf16* arow = hcur + (size_t)(mtile * 16 + (lane & 15)) * Hn;
      const __bf16* bg0 = whh + (size_t)(nb)          * Hn;
      const __bf16* bg1 = whh + (size_t)(nb + Hn)     * Hn;
      const __bf16* bg2 = whh + (size_t)(nb + 2 * Hn) * Hn;
      const __bf16* bg3 = whh + (size_t)(nb + 3 * Hn) * Hn;
      v8f acc0 = vzero8(), acc1 = vzero8(), acc2 = vzero8(), acc3 = vzero8();
      for (int k0 = 0; k0 < Hn; k0 += 32) {
        v16bf a = load_a_row(arow, half, k0);
        acc0 = wmma_bf(a, load_bt_row(bg0, half, k0), acc0);
        acc1 = wmma_bf(a, load_bt_row(bg1, half, k0), acc1);
        acc2 = wmma_bf(a, load_bt_row(bg2, half, k0), acc2);
        acc3 = wmma_bf(a, load_bt_row(bg3, half, k0), acc3);
      }
#pragma unroll
      for (int rr = 0; rr < 8; ++rr) {
        int b = mtile * 16 + rr + 8 * half;
        int hc = j * 16 + lcol;
        const float* xb = xgr + (size_t)b * Gn + hc;
        float gi = acc0[rr] + xb[0];
        float gf = acc1[rr] + xb[Hn];
        float gg = acc2[rr] + xb[2 * Hn];
        float go = acc3[rr] + xb[3 * Hn];
        float cc = sigm(gf) * cst[q][rr] + sigm(gi) * tanhf(gg);
        cst[q][rr] = cc;
        __bf16 hh = f2bf(sigm(go) * tanhf(cc));
        hnxt[(size_t)b * Hn + hc] = hh;
        outfeat[(size_t)(b * Wn + t) * (2 * Hn) + colOff + hc] = hh;
      }
    }
    __syncthreads();
  }
}

// ---------------- mean-pool over time + final tiny GEMM ----------------
__global__ void k_pool_out(const __bf16* __restrict__ feats2, const float* __restrict__ Wout,
                           float* __restrict__ out) {
  int b = blockIdx.x;
  int tid = threadIdx.x;                       // 256
  __shared__ float pooled[2 * Hn];
  __shared__ float red[256];
  for (int e = tid; e < 2 * Hn; e += blockDim.x) {
    float acc = 0.f;
    for (int t = 0; t < Wn; ++t) acc += bf2f(feats2[(size_t)(b * Wn + t) * (2 * Hn) + e]);
    pooled[e] = acc / (float)Wn;
  }
  __syncthreads();
  for (int j = 0; j < 4; ++j) {
    float p = 0.f;
    for (int e = tid; e < 2 * Hn; e += blockDim.x) p += pooled[e] * Wout[e * 4 + j];
    red[tid] = p;
    __syncthreads();
    for (int st = 128; st > 0; st >>= 1) { if (tid < st) red[tid] += red[tid + st]; __syncthreads(); }
    if (tid == 0) out[b * 4 + j] = red[0];
    __syncthreads();
  }
}

// ---------------- host orchestration ----------------
extern "C" void kernel_launch(void* const* d_in, const int* in_sizes, int n_in,
                              void* d_out, int out_size, void* d_ws, size_t ws_size,
                              hipStream_t stream) {
  (void)in_sizes; (void)n_in; (void)out_size; (void)ws_size;
  const int*   src        = (const int*)d_in[0];
  const int*   word_src   = (const int*)d_in[1];
  const float* char_table = (const float*)d_in[2];
  const float* word_table = (const float*)d_in[3];
  const float* w_bi  = (const float*)d_in[4];
  const float* b_bi  = (const float*)d_in[5];
  const float* w_tri = (const float*)d_in[6];
  const float* b_tri = (const float*)d_in[7];
  const float* Wa    = (const float*)d_in[8];
  const float* ba    = (const float*)d_in[9];
  const float* ua    = (const float*)d_in[10];
  const float* W1    = (const float*)d_in[11];
  const float* wih0  = (const float*)d_in[12];
  const float* whh0  = (const float*)d_in[13];
  const float* b0    = (const float*)d_in[14];
  const float* wih1  = (const float*)d_in[15];
  const float* whh1  = (const float*)d_in[16];
  const float* b1    = (const float*)d_in[17];
  const float* Wout  = (const float*)d_in[18];
  float* out = (float*)d_out;

  char* w = (char*)d_ws;
  size_t off = 0;
  auto alloc = [&](size_t bytes) -> void* {
    void* p = w + off; off += (bytes + 255) & ~(size_t)255; return p;
  };
  __bf16* Bbi    = (__bf16*)alloc((size_t)2 * Dn * Dn * 2);      // (256, 2*256) N-major
  __bf16* Btri   = (__bf16*)alloc((size_t)3 * Dn * Dn * 2);      // (256, 3*256)
  __bf16* WaT    = (__bf16*)alloc((size_t)(2*Dn) * (2*Dn) * 2);  // (512, 512) N-major
  __bf16* W1T    = (__bf16*)alloc((size_t)(2*Dn) * Dn * 2);      // (256, 512) N-major
  __bf16* wih0B  = (__bf16*)alloc((size_t)2 * Gn * (2*Dn) * 2);  // (G, 512) already N-major
  __bf16* whh0B  = (__bf16*)alloc((size_t)2 * Gn * Hn * 2);      // (G, 512)
  __bf16* wih1B  = (__bf16*)alloc((size_t)2 * Gn * (2*Hn) * 2);  // (G, 1024)
  __bf16* whh1B  = (__bf16*)alloc((size_t)2 * Gn * Hn * 2);      // (G, 512)
  __bf16* xpad   = (__bf16*)alloc((size_t)NW * (Cn + 2) * Dn * 2);
  __bf16* cat    = (__bf16*)alloc((size_t)NPOS * (2*Dn) * 2);
  float*  sbuf   = (float*) alloc((size_t)NPOS * 4);
  __bf16* wembB  = (__bf16*)alloc((size_t)NW * (2*Dn) * 2);
  __bf16* feats  = (__bf16*)alloc((size_t)NW * (2*Dn) * 2);
  float*  xg     = (float*) alloc((size_t)2 * NW * Gn * 4);      // reused by both layers
  __bf16* feats1 = (__bf16*)alloc((size_t)NW * (2*Hn) * 2);
  __bf16* feats2 = (__bf16*)alloc((size_t)NW * (2*Hn) * 2);

  // ---- stage 0: weight conversion to bf16, N-major (K-contiguous) layouts ----
  k_conv_w<<<(Dn*Dn*2 + 255)/256, 256, 0, stream>>>(w_bi,  Bbi,  2);
  k_conv_w<<<(Dn*Dn*3 + 255)/256, 256, 0, stream>>>(w_tri, Btri, 3);
  k_mat_bf<<<((2*Dn)*(2*Dn) + 255)/256, 256, 0, stream>>>(Wa, WaT, 2*Dn, 2*Dn, 1);
  k_mat_bf<<<((2*Dn)*Dn + 255)/256,     256, 0, stream>>>(W1, W1T, 2*Dn, Dn, 1);
  k_mat_bf<<<(2*Gn*(2*Dn) + 255)/256, 256, 0, stream>>>(wih0, wih0B, 2*Gn, 2*Dn, 0);
  k_mat_bf<<<(2*Gn*Hn + 255)/256,     256, 0, stream>>>(whh0, whh0B, 2*Gn, Hn,   0);
  k_mat_bf<<<(2*Gn*(2*Hn) + 255)/256, 256, 0, stream>>>(wih1, wih1B, 2*Gn, 2*Hn, 0);
  k_mat_bf<<<(2*Gn*Hn + 255)/256,     256, 0, stream>>>(whh1, whh1B, 2*Gn, Hn,   0);

  // ---- stage 1: char embedding gather (zero-padded) ----
  k_embed<<<NW * (Cn + 2), 256, 0, stream>>>(src, char_table, xpad);

  // ---- stage 2: bigram/trigram conv GEMMs (+bias +PE) -> cat ----
  k_conv_gemm<<<NPOS/32, 256, 0, stream>>>(xpad, Bbi,  b_bi,  2, cat, 0);
  k_conv_gemm<<<NPOS/32, 256, 0, stream>>>(xpad, Btri, b_tri, 3, cat, Dn);

  // ---- stage 3: attention scores, softmax, weighted sum ----
  k_attn_score<<<NPOS/16/8, 256, 0, stream>>>(cat, WaT, ba, ua, sbuf);
  k_attn_pool<<<NW, 128, 0, stream>>>(cat, sbuf, wembB);

  // ---- stage 4: feats = [wemb @ W1, word_table[word_src]] ----
  k_w1_gemm<<<NW/32, 256, 0, stream>>>(wembB, W1T, feats);
  k_word_gather<<<NW, 256, 0, stream>>>(word_src, word_table, feats);

  // ---- stage 5: BiLSTM layer 0 ----
  for (int d = 0; d < 2; ++d)
    k_inproj<<<dim3(NW/32, Gn/32/8), 256, 0, stream>>>(feats, 2*Dn,
        wih0B + (size_t)d*Gn*(2*Dn), b0 + d*Gn, xg + (size_t)d*NW*Gn);
  k_lstm<<<2, 1024, 2*Bn*Hn*sizeof(__bf16), stream>>>(xg, whh0B, feats1);

  // ---- stage 6: BiLSTM layer 1 ----
  for (int d = 0; d < 2; ++d)
    k_inproj<<<dim3(NW/32, Gn/32/8), 256, 0, stream>>>(feats1, 2*Hn,
        wih1B + (size_t)d*Gn*(2*Hn), b1 + d*Gn, xg + (size_t)d*NW*Gn);
  k_lstm<<<2, 1024, 2*Bn*Hn*sizeof(__bf16), stream>>>(xg, whh1B, feats2);

  // ---- stage 7: mean pool + output projection ----
  k_pool_out<<<Bn, 256, 0, stream>>>(feats2, Wout, out);
}
